// TSLR_36438502539713
// MI455X (gfx1250) — matrix operand order repaired and checked by reading, not compiled, not run on hardware
//
#include <hip/hip_runtime.h>

typedef __bf16 bf16;
typedef __attribute__((ext_vector_type(16))) __bf16 v16bf;
typedef __attribute__((ext_vector_type(8)))  __bf16 v8bf;
typedef __attribute__((ext_vector_type(8)))  float  v8f;

#define EPI_NONE        0
#define EPI_SIGMOID     1
#define EPI_RELUSQ_BF16 2
#define EPI_RESID       3
#define EPI_RESID_MUL   4

// ---------------------------------------------------------------------------
// Tiled f32 transpose, per batch: in [B][rows][cols] -> out [B][cols][rows]
// grid (cols/32, rows/32, B), block (32,8)
// ---------------------------------------------------------------------------
__global__ __launch_bounds__(256)
void vrwkv_transpose_f32(const float* __restrict__ in, float* __restrict__ out,
                         int rows, int cols) {
  __shared__ float tile[32][33];
  const int b = blockIdx.z;
  const float* pin  = in  + (size_t)b * rows * cols;
  float*       pout = out + (size_t)b * rows * cols;
  const int x  = blockIdx.x * 32 + threadIdx.x;
  const int y0 = blockIdx.y * 32;
#pragma unroll
  for (int i = 0; i < 32; i += 8)
    tile[threadIdx.y + i][threadIdx.x] = pin[(size_t)(y0 + threadIdx.y + i) * cols + x];
  __syncthreads();
  const int xr = blockIdx.y * 32 + threadIdx.x;  // row index in input
#pragma unroll
  for (int i = 0; i < 32; i += 8)
    pout[(size_t)(blockIdx.x * 32 + threadIdx.y + i) * rows + xr] = tile[threadIdx.x][threadIdx.y + i];
}

// ---------------------------------------------------------------------------
// OmniShift reparameterization: eff5x5[c] = a0*I + a1*w1 + a2*w3(pad) + a3*w5
// ---------------------------------------------------------------------------
__global__ void vrwkv_fold_conv(const float* __restrict__ alpha,
                                const float* __restrict__ w1,
                                const float* __restrict__ w3,
                                const float* __restrict__ w5,
                                float* __restrict__ eff, int C) {
  int i = blockIdx.x * blockDim.x + threadIdx.x;
  if (i >= C * 25) return;
  int c = i / 25, j = i % 25, r = j / 5, s = j % 5;
  float v = alpha[3] * w5[c * 25 + j];
  if (r >= 1 && r <= 3 && s >= 1 && s <= 3) v += alpha[2] * w3[c * 9 + (r - 1) * 3 + (s - 1)];
  if (r == 2 && s == 2) v += alpha[1] * w1[c] + alpha[0];
  eff[i] = v;
}

__global__ void vrwkv_cvt_bf16(const float* __restrict__ in, bf16* __restrict__ out, size_t n) {
  size_t i = (size_t)blockIdx.x * blockDim.x + threadIdx.x;
  if (i < n) out[i] = (bf16)in[i];
}

// ---------------------------------------------------------------------------
// LayerNorm over C per token; one wave32 per row; writes bf16.
// grid = M/4, block = 128 (4 waves)
// ---------------------------------------------------------------------------
__global__ __launch_bounds__(128)
void vrwkv_ln_bf16(const float* __restrict__ in, const float* __restrict__ g,
                   const float* __restrict__ b, bf16* __restrict__ out, int C) {
  const int wv = threadIdx.x >> 5, lane = threadIdx.x & 31;
  const size_t row = (size_t)blockIdx.x * 4 + wv;
  const float* p = in + row * C;
  float s = 0.f, s2 = 0.f;
  for (int c = lane; c < C; c += 32) { float x = p[c]; s += x; s2 += x * x; }
#pragma unroll
  for (int off = 16; off; off >>= 1) {
    s  += __shfl_xor(s,  off, 32);
    s2 += __shfl_xor(s2, off, 32);
  }
  const float mean = s / C;
  const float var  = s2 / C - mean * mean;
  const float inv  = rsqrtf(var + 1e-5f);
  bf16* q = out + row * C;
  for (int c = lane; c < C; c += 32)
    q[c] = (bf16)((p[c] - mean) * inv * g[c] + b[c]);
}

// ---------------------------------------------------------------------------
// Depthwise 5x5 conv (SAME, zero pad) in token-major [B,T,C] layout, bf16.
// ---------------------------------------------------------------------------
__global__ __launch_bounds__(256)
void vrwkv_dwconv5(const bf16* __restrict__ in, const float* __restrict__ eff,
                   bf16* __restrict__ out, int Bn, int C, int H, int W) {
  size_t i = (size_t)blockIdx.x * blockDim.x + threadIdx.x;
  size_t total = (size_t)Bn * H * W * C;
  if (i >= total) return;
  int c = (int)(i % C);
  size_t t = (i / C) % ((size_t)H * W);
  int b = (int)(i / ((size_t)C * H * W));
  int h = (int)(t / W), w = (int)(t % W);
  const bf16* base = in + (size_t)b * H * W * C;
  float acc = 0.f;
#pragma unroll
  for (int r = 0; r < 5; r++) {
    int hh = h + r - 2;
    if (hh < 0 || hh >= H) continue;
#pragma unroll
    for (int s = 0; s < 5; s++) {
      int ww = w + s - 2;
      if (ww < 0 || ww >= W) continue;
      acc += eff[c * 25 + r * 5 + s] * (float)base[((size_t)hh * W + ww) * C + c];
    }
  }
  out[i] = (bf16)acc;
}

// ---------------------------------------------------------------------------
// bf16 WMMA GEMM: out[m,n] = epi( sum_k A[m,k]*W[n,k] )
// A [M,K] bf16 row-major, W [N,K] bf16 row-major.
// Block = 256 thr = 8 waves (4 M-waves x 2 N-waves); wave tile = 32M x 64N
// (2 A-fragments share each B-fragment -> 0.75 KB loaded per WMMA).
// grid = (N/128, M/128). Requires M%128==0, N%128==0, K%32==0.
// ---------------------------------------------------------------------------
template <int EPI>
__global__ __launch_bounds__(256)
void vrwkv_gemm_wmma(const bf16* __restrict__ A, const bf16* __restrict__ Wm,
                     void* Out, const float* resid, const float* __restrict__ gamma,
                     const float* __restrict__ mul, int M, int N, int K) {
  const int tid   = threadIdx.x;
  const int lane  = tid & 31;
  const int wv    = tid >> 5;
  const int waveM = wv & 3;
  const int waveN = wv >> 2;
  const int mBase = blockIdx.y * 128 + waveM * 32;
  const int nBase = blockIdx.x * 128 + waveN * 64;
  const int l15   = lane & 15;
  const int lhi   = lane >> 4;

  const v8f vzero = {0.f, 0.f, 0.f, 0.f, 0.f, 0.f, 0.f, 0.f};
  v8f acc[2][4];
#pragma unroll
  for (int mi = 0; mi < 2; mi++)
#pragma unroll
    for (int j = 0; j < 4; j++) acc[mi][j] = vzero;

  // A lanes: rows mBase+l15 and mBase+16+l15; elements 0..7 <- K[kHalf*16..+7],
  // elements 8..15 <- K[kHalf*16+8..+7]
  const bf16* ap0 = A + (size_t)(mBase + l15) * K + lhi * 16;
  const bf16* ap1 = ap0 + (size_t)16 * K;
  // B lanes: col n = nBase+j*16+l15; contiguous 16 K-values starting at kHalf*16
  const bf16* bp[4];
#pragma unroll
  for (int j = 0; j < 4; j++)
    bp[j] = Wm + (size_t)(nBase + j * 16 + l15) * K + lhi * 16;

  for (int k0 = 0; k0 < K; k0 += 32) {
    __builtin_prefetch(ap0 + k0 + 128, 0, 1);
    __builtin_prefetch(ap1 + k0 + 128, 0, 1);
    v8bf lo0 = *(const v8bf*)(ap0 + k0);
    v8bf hi0 = *(const v8bf*)(ap0 + k0 + 8);
    v8bf lo1 = *(const v8bf*)(ap1 + k0);
    v8bf hi1 = *(const v8bf*)(ap1 + k0 + 8);
    v16bf a0, a1;
#pragma unroll
    for (int i = 0; i < 8; i++) {
      a0[i] = lo0[i]; a0[i + 8] = hi0[i];
      a1[i] = lo1[i]; a1[i + 8] = hi1[i];
    }
#pragma unroll
    for (int j = 0; j < 4; j++) {
      v16bf bfr = *(const v16bf*)(bp[j] + k0);
      acc[0][j] = __builtin_amdgcn_wmma_f32_16x16x32_bf16(
          false, a0, false, bfr, (short)0, acc[0][j], false, false);
      acc[1][j] = __builtin_amdgcn_wmma_f32_16x16x32_bf16(
          false, a1, false, bfr, (short)0, acc[1][j], false, false);
    }
  }

#pragma unroll
  for (int mi = 0; mi < 2; mi++) {
#pragma unroll
    for (int j = 0; j < 4; j++) {
      const int col = nBase + j * 16 + l15;
#pragma unroll
      for (int r = 0; r < 8; r++) {
        const int row = mBase + mi * 16 + r + lhi * 8;
        const size_t idx = (size_t)row * N + col;
        const float v = acc[mi][j][r];
        if (EPI == EPI_NONE) {
          ((float*)Out)[idx] = v;
        } else if (EPI == EPI_SIGMOID) {
          ((float*)Out)[idx] = 1.0f / (1.0f + __expf(-v));
        } else if (EPI == EPI_RELUSQ_BF16) {
          float t = v > 0.f ? v : 0.f;
          ((bf16*)Out)[idx] = (bf16)(t * t);
        } else if (EPI == EPI_RESID) {
          ((float*)Out)[idx] = resid[idx] + gamma[col] * v;
        } else if (EPI == EPI_RESID_MUL) {
          ((float*)Out)[idx] = resid[idx] + gamma[col] * (mul[idx] * v);
        }
      }
    }
  }
}

// ---------------------------------------------------------------------------
// RWKV WKV scans. One thread per (b,c); c contiguous across threads.
// ---------------------------------------------------------------------------
__global__ void vrwkv_wkv_row(const float* __restrict__ k, const float* __restrict__ v,
                              const float* __restrict__ decay, const float* __restrict__ first,
                              float* __restrict__ y, int Bn, int T, int C) {
  int gid = blockIdx.x * blockDim.x + threadIdx.x;
  if (gid >= Bn * C) return;
  int c = gid % C, b = gid / C;
  const float w = decay[c] / (float)T;
  const float u = first[c] / (float)T;
  float p = 0.f, q = 0.f, o = -1e30f;
  const size_t base = (size_t)b * T * C + c;
  for (int t = 0; t < T; t++) {
    size_t idx = base + (size_t)t * C;
    float kt = k[idx], vt = v[idx];
    float no = fmaxf(o, u + kt);
    float Ae = __expf(o - no), Be = __expf(u + kt - no);
    y[idx] = (Ae * p + Be * vt) / (Ae * q + Be);
    float no2 = fmaxf(w + o, kt);
    float A2 = __expf(w + o - no2), B2 = __expf(kt - no2);
    p = A2 * p + B2 * vt;
    q = A2 * q + B2;
    o = no2;
  }
}

__global__ void vrwkv_wkv_col(const float* __restrict__ k, const float* __restrict__ v,
                              const float* __restrict__ decay, const float* __restrict__ first,
                              float* __restrict__ y, int Bn, int H, int W, int C) {
  int gid = blockIdx.x * blockDim.x + threadIdx.x;
  if (gid >= Bn * C) return;
  int c = gid % C, b = gid / C;
  const int T = H * W;
  const float w = decay[c] / (float)T;
  const float u = first[c] / (float)T;
  float p = 0.f, q = 0.f, o = -1e30f;
  const size_t base = (size_t)b * T * C + c;
  for (int ww = 0; ww < W; ww++) {
    for (int hh = 0; hh < H; hh++) {
      size_t idx = base + ((size_t)hh * W + ww) * C;
      float kt = k[idx], vt = v[idx];
      float no = fmaxf(o, u + kt);
      float Ae = __expf(o - no), Be = __expf(u + kt - no);
      y[idx] = (Ae * p + Be * vt) / (Ae * q + Be);
      float no2 = fmaxf(w + o, kt);
      float A2 = __expf(w + o - no2), B2 = __expf(kt - no2);
      p = A2 * p + B2 * vt;
      q = A2 * q + B2;
      o = no2;
    }
  }
}

__global__ void vrwkv_mul_cvt(const float* __restrict__ a, const float* __restrict__ b,
                              bf16* __restrict__ out, size_t n) {
  size_t i = (size_t)blockIdx.x * blockDim.x + threadIdx.x;
  if (i < n) out[i] = (bf16)(a[i] * b[i]);
}

// ---------------------------------------------------------------------------
extern "C" void kernel_launch(void* const* d_in, const int* in_sizes, int n_in,
                              void* d_out, int out_size, void* d_ws, size_t ws_size,
                              hipStream_t stream) {
  (void)in_sizes; (void)n_in; (void)out_size; (void)ws_size;
  const int Bn = 8, C = 384, H = 64, Wd = 64, T = H * Wd, HID = 4 * C;
  const size_t M = (size_t)Bn * T;  // 32768 tokens

  const float* x         = (const float*)d_in[0];
  const float* ln1_g     = (const float*)d_in[1];
  const float* ln1_b     = (const float*)d_in[2];
  const float* ln2_g     = (const float*)d_in[3];
  const float* ln2_b     = (const float*)d_in[4];
  const float* gamma1    = (const float*)d_in[5];
  const float* gamma2    = (const float*)d_in[6];
  const float* att_alpha = (const float*)d_in[7];
  const float* att_w1    = (const float*)d_in[8];
  const float* att_w3    = (const float*)d_in[9];
  const float* att_w5    = (const float*)d_in[10];
  const float* att_wk    = (const float*)d_in[11];
  const float* att_wv    = (const float*)d_in[12];
  const float* att_wr    = (const float*)d_in[13];
  const float* att_wo    = (const float*)d_in[14];
  const float* sp_decay  = (const float*)d_in[15];
  const float* sp_first  = (const float*)d_in[16];
  const float* ffn_alpha = (const float*)d_in[17];
  const float* ffn_w1    = (const float*)d_in[18];
  const float* ffn_w3    = (const float*)d_in[19];
  const float* ffn_w5    = (const float*)d_in[20];
  const float* ffn_wk    = (const float*)d_in[21];
  const float* ffn_wv    = (const float*)d_in[22];
  const float* ffn_wr    = (const float*)d_in[23];

  // ---- workspace carve ----
  char* ws = (char*)d_ws;
  size_t off = 0;
  auto take = [&](size_t bytes) -> char* {
    char* p = ws + off;
    off = (off + bytes + 255) & ~(size_t)255;
    return p;
  };
  float* tok  = (float*)take(M * C * sizeof(float));        // residual state [M,C]
  bf16*  xn   = (bf16*) take(M * C * sizeof(bf16));         // LN output
  bf16*  xa   = (bf16*) take(M * C * sizeof(bf16));         // conv output / GEMM A
  float* kb   = (float*)take(M * C * sizeof(float));        // k
  float* vb   = (float*)take(M * C * sizeof(float));        // v, later y2
  float* rb   = (float*)take(M * C * sizeof(float));        // sigmoid(r)
  float* y1   = (float*)take(M * C * sizeof(float));        // row-scan output
  bf16*  kkb  = (bf16*) take(M * (size_t)HID * sizeof(bf16)); // relu^2 [M,HID]
  bf16*  wkA  = (bf16*) take((size_t)C * C * sizeof(bf16));
  bf16*  wvA  = (bf16*) take((size_t)C * C * sizeof(bf16));
  bf16*  wrA  = (bf16*) take((size_t)C * C * sizeof(bf16));
  bf16*  woA  = (bf16*) take((size_t)C * C * sizeof(bf16));
  bf16*  wkF  = (bf16*) take((size_t)HID * C * sizeof(bf16));
  bf16*  wvF  = (bf16*) take((size_t)C * HID * sizeof(bf16));
  bf16*  wrF  = (bf16*) take((size_t)C * C * sizeof(bf16));
  float* effA = (float*)take((size_t)C * 25 * sizeof(float));
  float* effF = (float*)take((size_t)C * 25 * sizeof(float));

  const dim3 tb(32, 8);
  const unsigned ewGrid = (unsigned)((M * C + 255) / 256);

  // 0) NCHW -> [B,T,C]
  vrwkv_transpose_f32<<<dim3(T / 32, C / 32, Bn), tb, 0, stream>>>(x, tok, C, T);

  // 1) fold OmniShift weights, convert GEMM weights to bf16
  vrwkv_fold_conv<<<(C * 25 + 255) / 256, 256, 0, stream>>>(att_alpha, att_w1, att_w3, att_w5, effA, C);
  vrwkv_fold_conv<<<(C * 25 + 255) / 256, 256, 0, stream>>>(ffn_alpha, ffn_w1, ffn_w3, ffn_w5, effF, C);
  auto cvt = [&](const float* src, bf16* dst, size_t n) {
    vrwkv_cvt_bf16<<<(unsigned)((n + 255) / 256), 256, 0, stream>>>(src, dst, n);
  };
  cvt(att_wk, wkA, (size_t)C * C);
  cvt(att_wv, wvA, (size_t)C * C);
  cvt(att_wr, wrA, (size_t)C * C);
  cvt(att_wo, woA, (size_t)C * C);
  cvt(ffn_wk, wkF, (size_t)HID * C);
  cvt(ffn_wv, wvF, (size_t)C * HID);
  cvt(ffn_wr, wrF, (size_t)C * C);

  // ---- spatial mix ----
  vrwkv_ln_bf16<<<(unsigned)(M / 4), 128, 0, stream>>>(tok, ln1_g, ln1_b, xn, C);
  vrwkv_dwconv5<<<ewGrid, 256, 0, stream>>>(xn, effA, xa, Bn, C, H, Wd);

  const dim3 gGridC(C / 128, (unsigned)(M / 128));    // N=384
  const dim3 gGridH(HID / 128, (unsigned)(M / 128));  // N=1536
  vrwkv_gemm_wmma<EPI_NONE><<<gGridC, 256, 0, stream>>>(xa, wkA, kb, nullptr, nullptr, nullptr, (int)M, C, C);
  vrwkv_gemm_wmma<EPI_NONE><<<gGridC, 256, 0, stream>>>(xa, wvA, vb, nullptr, nullptr, nullptr, (int)M, C, C);
  vrwkv_gemm_wmma<EPI_SIGMOID><<<gGridC, 256, 0, stream>>>(xa, wrA, rb, nullptr, nullptr, nullptr, (int)M, C, C);

  vrwkv_wkv_row<<<(Bn * C + 255) / 256, 256, 0, stream>>>(kb, vb, sp_decay, sp_first, y1, Bn, T, C);
  vrwkv_wkv_col<<<(Bn * C + 255) / 256, 256, 0, stream>>>(kb, y1, sp_decay + C, sp_first + C, vb, Bn, H, Wd, C);

  vrwkv_mul_cvt<<<ewGrid, 256, 0, stream>>>(rb, vb, xa, M * C);
  vrwkv_gemm_wmma<EPI_RESID><<<gGridC, 256, 0, stream>>>(xa, woA, tok, tok, gamma1, nullptr, (int)M, C, C);

  // ---- channel mix ----
  vrwkv_ln_bf16<<<(unsigned)(M / 4), 128, 0, stream>>>(tok, ln2_g, ln2_b, xn, C);
  vrwkv_dwconv5<<<ewGrid, 256, 0, stream>>>(xn, effF, xa, Bn, C, H, Wd);

  vrwkv_gemm_wmma<EPI_RELUSQ_BF16><<<gGridH, 256, 0, stream>>>(xa, wkF, kkb, nullptr, nullptr, nullptr, (int)M, HID, C);
  vrwkv_gemm_wmma<EPI_SIGMOID><<<gGridC, 256, 0, stream>>>(xa, wrF, rb, nullptr, nullptr, nullptr, (int)M, C, C);
  vrwkv_gemm_wmma<EPI_RESID_MUL><<<gGridC, 256, 0, stream>>>(kkb, wvF, tok, tok, gamma2, rb, (int)M, C, HID);

  // 9) [B,T,C] -> NCHW output
  vrwkv_transpose_f32<<<dim3(C / 32, T / 32, Bn), tb, 0, stream>>>(tok, (float*)d_out, T, C);
}
